// THAttentionBlock_15728170238613
// MI455X (gfx1250) — matrix operand order repaired and checked by reading, not compiled, not run on hardware
//
#include <hip/hip_runtime.h>
#include <hip/hip_bf16.h>
#include <math.h>

// ---------------------------------------------------------------------------
// Types for CDNA5 WMMA (wave32): v16bf A/B fragments, v8f accumulator.
// ---------------------------------------------------------------------------
typedef __attribute__((ext_vector_type(16))) __bf16 v16bf;
typedef __attribute__((ext_vector_type(8)))  __bf16 v8bf;
typedef __attribute__((ext_vector_type(8)))  float  v8f;

__device__ __forceinline__ __bf16 f2bf(float f) {
  unsigned int u = __builtin_bit_cast(unsigned int, f);
  unsigned int lsb = (u >> 16) & 1u;
  u += 0x7fffu + lsb;                       // round-to-nearest-even
  unsigned short h = (unsigned short)(u >> 16);
  return __builtin_bit_cast(__bf16, h);
}
__device__ __forceinline__ float bf2f(__bf16 b) {
  unsigned int u = ((unsigned int)__builtin_bit_cast(unsigned short, b)) << 16;
  return __builtin_bit_cast(float, u);
}
__device__ __forceinline__ float toF(float v)  { return v; }
__device__ __forceinline__ float toF(__bf16 v) { return bf2f(v); }

// ---------------------------------------------------------------------------
// CDNA5 async global->LDS copy (ASYNCcnt-tracked, no VGPR round trip).
// LDS operand is the DS-style byte address = low 32 bits of the flat address.
// ---------------------------------------------------------------------------
__device__ __forceinline__ void async_copy_b128(void* lds_ptr, const void* gptr) {
  unsigned lds_addr = (unsigned)(unsigned long long)lds_ptr;
  asm volatile("global_load_async_to_lds_b128 %0, %1, off"
               :: "v"(lds_addr), "v"(gptr)
               : "memory");
}
__device__ __forceinline__ void wait_asynccnt0() {
  asm volatile("s_wait_asynccnt 0x0" ::: "memory");
}

// ---------------------------------------------------------------------------
// f32 -> bf16 weight conversion
// ---------------------------------------------------------------------------
__global__ __launch_bounds__(256) void f32_to_bf16_kernel(
    const float* __restrict__ in, __bf16* __restrict__ out, int n) {
  int i = blockIdx.x * 256 + threadIdx.x;
  if (i < n) out[i] = f2bf(in[i]);
}

// ---------------------------------------------------------------------------
// LayerNorm (population var, eps=1e-5), one block per row, bf16 output.
// ---------------------------------------------------------------------------
template <typename TIN>
__global__ __launch_bounds__(256) void layernorm_to_bf16_kernel(
    const TIN* __restrict__ x, const float* __restrict__ g,
    const float* __restrict__ b, __bf16* __restrict__ out, int C) {
  __shared__ float red[256];
  const int row = blockIdx.x;
  const TIN* xr = x + (size_t)row * C;
  float s = 0.f, ss = 0.f;
  for (int i = threadIdx.x; i < C; i += 256) {
    float v = toF(xr[i]);
    s += v; ss += v * v;
  }
  red[threadIdx.x] = s; __syncthreads();
  for (int o = 128; o > 0; o >>= 1) {
    if (threadIdx.x < o) red[threadIdx.x] += red[threadIdx.x + o];
    __syncthreads();
  }
  const float mean = red[0] / (float)C;
  __syncthreads();
  red[threadIdx.x] = ss; __syncthreads();
  for (int o = 128; o > 0; o >>= 1) {
    if (threadIdx.x < o) red[threadIdx.x] += red[threadIdx.x + o];
    __syncthreads();
  }
  const float var = red[0] / (float)C - mean * mean;
  const float inv = rsqrtf(var + 1e-5f);
  __bf16* orow = out + (size_t)row * C;
  for (int i = threadIdx.x; i < C; i += 256)
    orow[i] = f2bf((toF(xr[i]) - mean) * inv * g[i] + b[i]);
}

// ---------------------------------------------------------------------------
// WMMA bf16 GEMM:  out[m,n] = epilogue( sum_k A[m,k]*W[n,k] + bias[n] )
//   A: M x K row-major bf16 ;  W: N x K row-major bf16 (i.e. B^T, native)
// Block: 256 threads (8 waves), tile 128x128. K staged 32 deep into LDS via
// GLOBAL_LOAD_ASYNC_TO_LDS_B128, DOUBLE-BUFFERED: copies for tile kt+1 run
// concurrently with WMMA compute on tile kt; one barrier per K-step.
// Wave grid 2x4; each wave owns 64x32 = 4x2 v_wmma_f32_16x16x32_bf16 tiles.
// ---------------------------------------------------------------------------
#define BM 128
#define BN 128
#define BK 32

enum { EPI_RES1 = 0, EPI_GELU = 1, EPI_RES2 = 2 };

template <int EPI>
__global__ __launch_bounds__(256) void gemm_wmma_bf16_kernel(
    const __bf16* __restrict__ A, const __bf16* __restrict__ W,
    const float* __restrict__ bias, const float* __restrict__ gamma,
    const float* __restrict__ resid, float* __restrict__ outF,
    __bf16* __restrict__ outB, int M, int N, int K) {
  __shared__ __bf16 sA[2][BM * BK];   // 2 x 8 KB
  __shared__ __bf16 sB[2][BN * BK];   // 2 x 8 KB

  const int tid  = threadIdx.x;
  const int lane = tid & 31;
  const int wave = tid >> 5;
  const int wr   = wave >> 2;   // 0..1 -> 64-row strip
  const int wc   = wave & 3;    // 0..3 -> 32-col strip
  const int ln15 = lane & 15;
  const int kh   = lane >> 4;   // half-selector per documented VGPR layouts

  const int bM = blockIdx.y * BM;
  const int bN = blockIdx.x * BN;

  // Per-thread staging geometry: tiles are 128x32 bf16 = 512 x 16B chunks,
  // 2 chunks per thread per matrix; 4 chunks per row (kc), row index m.
  const int m0 = tid >> 2,        kc0 = tid & 3;
  const int m1 = (256 + tid) >> 2, kc1 = (256 + tid) & 3;
  const __bf16* aP0 = A + (size_t)(bM + m0) * K + kc0 * 8;
  const __bf16* aP1 = A + (size_t)(bM + m1) * K + kc1 * 8;
  const __bf16* bP0 = W + (size_t)(bN + m0) * K + kc0 * 8;
  const __bf16* bP1 = W + (size_t)(bN + m1) * K + kc1 * 8;

  auto stage = [&](int buf, int k0) {
    async_copy_b128(&sA[buf][m0 * BK + kc0 * 8], aP0 + k0);
    async_copy_b128(&sB[buf][m0 * BK + kc0 * 8], bP0 + k0);
    async_copy_b128(&sA[buf][m1 * BK + kc1 * 8], aP1 + k0);
    async_copy_b128(&sB[buf][m1 * BK + kc1 * 8], bP1 + k0);
  };

  v8f acc[4][2] = {};

  const int nK = K / BK;
  stage(0, 0);  // prologue: tile 0 -> buffer 0

  for (int kt = 0; kt < nK; ++kt) {
    const int cur = kt & 1;
    wait_asynccnt0();   // this wave's copies into buf `cur` have landed
    __syncthreads();    // every wave's copies visible; prior reads drained
    if (kt + 1 < nK) stage(cur ^ 1, (kt + 1) * BK);  // overlap with compute
    if (kt + 2 < nK) {  // warm L2 for the tile after that
      const int k2 = (kt + 2) * BK;
      __builtin_prefetch(aP0 + k2, 0, 1);
      __builtin_prefetch(bP0 + k2, 0, 1);
      __builtin_prefetch(aP1 + k2, 0, 1);
      __builtin_prefetch(bP1 + k2, 0, 1);
    }

    // Build fragments per CDNA5 ISA layouts.
    // A (16-bit 16x32): lane L row = L&15; elems 0..7 -> K=8*kh..+7,
    //                   elems 8..15 -> K=16+8*kh..+7.
    v16bf aF[4], bF[2];
#pragma unroll
    for (int ti = 0; ti < 4; ++ti) {
      const int Mrow = wr * 64 + ti * 16 + ln15;
      const v8bf lo = *(const v8bf*)(&sA[cur][Mrow * BK + 8 * kh]);
      const v8bf hi = *(const v8bf*)(&sA[cur][Mrow * BK + 16 + 8 * kh]);
      v16bf f;
#pragma unroll
      for (int e = 0; e < 8; ++e) { f[e] = lo[e]; f[e + 8] = hi[e]; }
      aF[ti] = f;
    }
    // B (16-bit 32x16): lane L col = L&15; elems 0..7 -> K=8*kh..+7,
    //                   elems 8..15 -> K=16+8*kh..+7.
#pragma unroll
    for (int tj = 0; tj < 2; ++tj) {
      const int Ncol = wc * 32 + tj * 16 + ln15;
      const v8bf lo = *(const v8bf*)(&sB[cur][Ncol * BK + 8 * kh]);
      const v8bf hi = *(const v8bf*)(&sB[cur][Ncol * BK + 16 + 8 * kh]);
      v16bf f;
#pragma unroll
      for (int e = 0; e < 8; ++e) { f[e] = lo[e]; f[e + 8] = hi[e]; }
      bF[tj] = f;
    }
#pragma unroll
    for (int ti = 0; ti < 4; ++ti)
#pragma unroll
      for (int tj = 0; tj < 2; ++tj)
        acc[ti][tj] = __builtin_amdgcn_wmma_f32_16x16x32_bf16(
            false, aF[ti], false, bF[tj], (short)0, acc[ti][tj], false, false);
  }

  // Epilogue. C/D layout: VGPR r -> row = r + 8*kh, col = lane&15.
#pragma unroll
  for (int ti = 0; ti < 4; ++ti) {
#pragma unroll
    for (int tj = 0; tj < 2; ++tj) {
      const int nn = bN + wc * 32 + tj * 16 + ln15;
      const float bv = bias[nn];
#pragma unroll
      for (int r = 0; r < 8; ++r) {
        const int mm = bM + wr * 64 + ti * 16 + r + 8 * kh;
        const float v = acc[ti][tj][r] + bv;
        const size_t idx = (size_t)mm * N + nn;
        if (EPI == EPI_GELU) {
          const float gl = 0.5f * v * (1.0f + erff(v * 0.70710678118654752f));
          outB[idx] = f2bf(gl);
        } else {
          outF[idx] = resid[idx] + gamma[nn] * v;
        }
      }
    }
  }
}

// ---------------------------------------------------------------------------
// Orchestration. Live dataflow only (attention branch is dead code in ref).
// ---------------------------------------------------------------------------
extern "C" void kernel_launch(void* const* d_in, const int* in_sizes, int n_in,
                              void* d_out, int out_size, void* d_ws, size_t ws_size,
                              hipStream_t stream) {
  (void)in_sizes; (void)n_in; (void)out_size; (void)ws_size;

  const float* x      = (const float*)d_in[0];
  const float* ln1_g  = (const float*)d_in[1];
  const float* ln1_b  = (const float*)d_in[2];
  // d_in[3] qkv_w, d_in[6..9] pre/post talking-head: dead in reference.
  const float* proj_w = (const float*)d_in[4];
  const float* proj_b = (const float*)d_in[5];
  const float* ln2_g  = (const float*)d_in[10];
  const float* ln2_b  = (const float*)d_in[11];
  const float* fc1_w  = (const float*)d_in[12];
  const float* fc1_b  = (const float*)d_in[13];
  const float* lnh_g  = (const float*)d_in[14];
  const float* lnh_b  = (const float*)d_in[15];
  const float* fc2_w  = (const float*)d_in[16];
  const float* fc2_b  = (const float*)d_in[17];
  const float* gamma1 = (const float*)d_in[18];
  const float* gamma2 = (const float*)d_in[19];

  const int Bb = 16, Nn = 1024, C = 1024, Hh = 4096;
  const int M = Bb * Nn;  // 16384 rows

  // Workspace carve-up (256B aligned).
  char* ws = (char*)d_ws;
  size_t off = 0;
  auto carve = [&](size_t bytes) -> char* {
    char* p = ws + off;
    off += (bytes + 255) & ~(size_t)255;
    return p;
  };
  __bf16* projW_bf = (__bf16*)carve((size_t)C * C * 2);    //   2 MB
  __bf16* fc1W_bf  = (__bf16*)carve((size_t)Hh * C * 2);   //   8 MB
  __bf16* fc2W_bf  = (__bf16*)carve((size_t)C * Hh * 2);   //   8 MB
  __bf16* act_bf   = (__bf16*)carve((size_t)M * C * 2);    //  32 MB (y, then z)
  float*  x1       = (float*)carve((size_t)M * C * 4);     //  64 MB
  __bf16* h_bf     = (__bf16*)carve((size_t)M * Hh * 2);   // 128 MB

  // Weights -> bf16 once per launch.
  f32_to_bf16_kernel<<<(C * C + 255) / 256, 256, 0, stream>>>(proj_w, projW_bf, C * C);
  f32_to_bf16_kernel<<<(Hh * C + 255) / 256, 256, 0, stream>>>(fc1_w, fc1W_bf, Hh * C);
  f32_to_bf16_kernel<<<(C * Hh + 255) / 256, 256, 0, stream>>>(fc2_w, fc2W_bf, C * Hh);

  // y = LN(x) -> bf16
  layernorm_to_bf16_kernel<float><<<M, 256, 0, stream>>>(x, ln1_g, ln1_b, act_bf, C);

  // x1 = x + gamma1 * (y @ proj_w^T + proj_b)
  {
    dim3 grid(C / BN, M / BM);
    gemm_wmma_bf16_kernel<EPI_RES1><<<grid, 256, 0, stream>>>(
        act_bf, projW_bf, proj_b, gamma1, x, x1, nullptr, M, C, C);
  }

  // z = LN(x1) -> bf16 (reuse act_bf)
  layernorm_to_bf16_kernel<float><<<M, 256, 0, stream>>>(x1, ln2_g, ln2_b, act_bf, C);

  // h = gelu(z @ fc1_w^T + fc1_b) -> bf16
  {
    dim3 grid(Hh / BN, M / BM);
    gemm_wmma_bf16_kernel<EPI_GELU><<<grid, 256, 0, stream>>>(
        act_bf, fc1W_bf, fc1_b, nullptr, nullptr, nullptr, h_bf, M, Hh, C);
  }

  // h = LN(h) in place (each thread reads/writes only its own elements)
  layernorm_to_bf16_kernel<__bf16><<<M, 256, 0, stream>>>(h_bf, lnh_g, lnh_b, h_bf, Hh);

  // out = x1 + gamma2 * (h @ fc2_w^T + fc2_b)
  {
    dim3 grid(C / BN, M / BM);
    gemm_wmma_bf16_kernel<EPI_RES2><<<grid, 256, 0, stream>>>(
        h_bf, fc2W_bf, fc2_b, gamma2, x1, (float*)d_out, nullptr, M, C, Hh);
  }
}
